// AdaptiveGraphConv_81561428951413
// MI455X (gfx1250) — matrix operand order repaired and checked by reading, not compiled, not run on hardware
//
#include <hip/hip_runtime.h>
#include <stdint.h>

// Problem constants (from reference setup_inputs)
#define BB 32
#define TT 128
#define FF 32
#define HH 4
#define HDD 16
#define RR 16
#define NN (TT * FF)          // 4096
#define ALPHA 0.2f
#define THRESH 0.01f

typedef __attribute__((ext_vector_type(16))) _Float16 v16h;
typedef __attribute__((ext_vector_type(8)))  _Float16 v8h;
typedef __attribute__((ext_vector_type(8)))  float    v8f;

// ---------------- fragment builders (CDNA5 16-bit WMMA layouts) -------------
// A (16xK, 16-bit): lane L holds row M=L%16; here elem j -> K = (L/16)*16 + j (same order as B)
//   => two contiguous runs of 8 halves.
// B (Kx16, 16-bit): lane L holds col N=L%16; here elem j -> K = (L/16)*16 + j (this code's order, same as A)
//   => one contiguous run of 16 halves.
__device__ __forceinline__ v16h make_a_frag(const _Float16* run1, const _Float16* run2) {
  v8h lo = *(const v8h*)run1;
  v8h hi = *(const v8h*)run2;
  v16h a;
#pragma unroll
  for (int j = 0; j < 8; ++j) { a[j] = lo[j]; a[j + 8] = hi[j]; }
  return a;
}
__device__ __forceinline__ v16h make_b_frag(const _Float16* p) {
  v8h lo = *(const v8h*)p;
  v8h hi = *(const v8h*)(p + 8);
  v16h b;
#pragma unroll
  for (int j = 0; j < 8; ++j) { b[j] = lo[j]; b[j + 8] = hi[j]; }
  return b;
}

__device__ __forceinline__ float leaky(float z) { return z > 0.0f ? z : ALPHA * z; }

__device__ __forceinline__ float hash01(uint32_t x) {
  x ^= x >> 17; x *= 0xed5ad4bbu; x ^= x >> 11; x *= 0xac4c1b51u;
  x ^= x >> 15; x *= 0x31848babu; x ^= x >> 14;
  return (float)(x >> 8) * (1.0f / 16777216.0f);
}

// ---------------- kernel 1: feature attention per head ----------------------
// grid = H, block = 32 (one thread per row f1)
__global__ void fa_kernel(const float* __restrict__ ff,   // [H][2][F][R]
                          float* __restrict__ fa,         // [H][F][F]
                          _Float16* __restrict__ faT,     // [H][F][F] (transposed)
                          float* __restrict__ rs_fa) {    // [H][F]
  const int h = blockIdx.x, f1 = threadIdx.x;
  const float* U = ff + ((h * 2 + 0) * FF) * RR;
  const float* V = ff + ((h * 2 + 1) * FF) * RR;
  float row[FF];
  float m = -1e30f;
#pragma unroll 4
  for (int f2 = 0; f2 < FF; ++f2) {
    float s = 0.0f;
    for (int r = 0; r < RR; ++r) s += U[f1 * RR + r] * V[f2 * RR + r];
    row[f2] = leaky(s);
    m = fmaxf(m, row[f2]);
  }
  float den = 0.0f;
  for (int f2 = 0; f2 < FF; ++f2) { row[f2] = expf(row[f2] - m); den += row[f2]; }
  float inv = 1.0f / den;
  float rs = 0.0f;
  for (int f2 = 0; f2 < FF; ++f2) {
    float p = row[f2] * inv;
    float val = (p > THRESH && f2 != f1) ? p : 0.0f;
    fa[(h * FF + f1) * FF + f2] = val;
    faT[(h * FF + f2) * FF + f1] = (_Float16)val;
    rs += val;
  }
  rs_fa[h * FF + f1] = rs;
}

// ---------------- kernel 2: transpose weight[1] to half ---------------------
__global__ void w1t_kernel(const float* __restrict__ weight, _Float16* __restrict__ W1T) {
  int idx = blockIdx.x * blockDim.x + threadIdx.x;  // [H][HD][F]
  if (idx >= HH * HDD * FF) return;
  int h = idx / (HDD * FF);
  int r = idx % (HDD * FF);
  int j = r / FF, f = r % FF;
  W1T[idx] = (_Float16)weight[((HH + h) * FF + f) * HDD + j];  // k=1 slice
}

// ---------------- kernel 3: xp = x @ W0h, q, k -------------------------------
// one thread per (h,b,t)
__global__ void xp_kernel(const float* __restrict__ x,       // [B][T][F]
                          const float* __restrict__ weight,  // [2][H][F][HD]
                          const float* __restrict__ tq,      // [H][HD]
                          const float* __restrict__ tk,      // [H][HD]
                          float* __restrict__ xp,            // [H][B][T][HD]
                          float* __restrict__ q,             // [H][B][T]
                          float* __restrict__ kk) {          // [H][B][T]
  int gid = blockIdx.x * blockDim.x + threadIdx.x;
  if (gid >= HH * BB * TT) return;
  int h = gid / (BB * TT);
  int r = gid % (BB * TT);
  int b = r / TT, t = r % TT;
  const float* xr = x + (b * TT + t) * FF;
  float xv[FF];
#pragma unroll
  for (int f = 0; f < FF; ++f) xv[f] = xr[f];
  float qa = 0.0f, ka = 0.0f;
  for (int j = 0; j < HDD; ++j) {
    float s = 0.0f;
#pragma unroll 8
    for (int f = 0; f < FF; ++f) s += xv[f] * weight[(h * FF + f) * HDD + j];  // k=0 slice
    xp[((h * BB + b) * TT + t) * HDD + j] = s;
    qa += s * tq[h * HDD + j];
    ka += s * tk[h * HDD + j];
  }
  q[(h * BB + b) * TT + t] = qa;
  kk[(h * BB + b) * TT + t] = ka;
}

// ---------------- kernel 4: temporal attention rows --------------------------
// grid (T, H), block T=128, one thread per t2
__global__ void ta_kernel(const float* __restrict__ q, const float* __restrict__ kk,
                          float* __restrict__ ta,        // [H][T][T]
                          _Float16* __restrict__ taT,    // [H][T][T] transposed
                          float* __restrict__ rs_ta) {   // [H][T]
  const int h = blockIdx.y, t1 = blockIdx.x, t2 = threadIdx.x;
  __shared__ float red[TT];
  float s = 0.0f;
  for (int b = 0; b < BB; ++b) {
    float z = q[(h * BB + b) * TT + t1] + kk[(h * BB + b) * TT + t2];
    s += leaky(z);
  }
  s *= (1.0f / BB);
  red[t2] = s; __syncthreads();
  for (int st = TT / 2; st > 0; st >>= 1) {
    if (t2 < st) red[t2] = fmaxf(red[t2], red[t2 + st]);
    __syncthreads();
  }
  float m = red[0]; __syncthreads();
  float e = expf(s - m);
  red[t2] = e; __syncthreads();
  for (int st = TT / 2; st > 0; st >>= 1) {
    if (t2 < st) red[t2] += red[t2 + st];
    __syncthreads();
  }
  float p = e / red[0]; __syncthreads();
  float val = (p > THRESH && t2 != t1) ? p : 0.0f;
  ta[(h * TT + t1) * TT + t2] = val;
  taT[(h * TT + t2) * TT + t1] = (_Float16)val;
  red[t2] = val; __syncthreads();
  for (int st = TT / 2; st > 0; st >>= 1) {
    if (t2 < st) red[t2] += red[t2 + st];
    __syncthreads();
  }
  if (t2 == 0) rs_ta[h * TT + t1] = red[0];
}

// ---------------- kernel 5: dinv + power iteration ---------------------------
// grid = H, block = 256 (each thread owns 16 of the 4096 nodes)
__global__ void power_kernel(const float* __restrict__ ta, const float* __restrict__ fa,
                             const float* __restrict__ rs_ta, const float* __restrict__ rs_fa,
                             float* __restrict__ dinv,   // [H][T][F]
                             float* __restrict__ cscale) // [H]
{
  const int h = blockIdx.x, tid = threadIdx.x;
  __shared__ float v[NN];
  __shared__ float w[NN];
  __shared__ float red[256];
  float dloc[16];
#pragma unroll
  for (int i = 0; i < 16; ++i) {
    int idx = tid + 256 * i;
    int t = idx >> 5, f = idx & 31;
    float d = rs_ta[h * TT + t] + rs_fa[h * FF + f] + 1.0f;
    float di = rsqrtf(d + 1e-10f);
    dloc[i] = di;
    dinv[(h * TT + t) * FF + f] = di;
    uint32_t s0 = (uint32_t)(idx * 2 + h * 100003 + 12345);
    float u1 = fmaxf(hash01(s0), 1e-7f);
    float u2 = hash01(s0 + 1u);
    v[idx] = sqrtf(-2.0f * logf(u1)) * cosf(6.2831853f * u2);
  }
  __syncthreads();
  float lv[16];
  for (int it = 0; it < 3; ++it) {
#pragma unroll
    for (int i = 0; i < 16; ++i) { int idx = tid + 256 * i; w[idx] = v[idx] * dloc[i]; }
    __syncthreads();
    float loc = 0.0f;
    for (int i = 0; i < 16; ++i) {
      int idx = tid + 256 * i;
      int t = idx >> 5, f = idx & 31;
      float s = w[idx];
      const float* tarow = ta + (h * TT + t) * TT;
      for (int t2 = 0; t2 < TT; ++t2) s += tarow[t2] * w[t2 * FF + f];
      const float* farow = fa + (h * FF + f) * FF;
      const float* wrow = w + t * FF;
      for (int f2 = 0; f2 < FF; ++f2) s += farow[f2] * wrow[f2];
      lv[i] = v[idx] - dloc[i] * s;
      loc += lv[i] * lv[i];
    }
    red[tid] = loc; __syncthreads();
    for (int st = 128; st > 0; st >>= 1) {
      if (tid < st) red[tid] += red[tid + st];
      __syncthreads();
    }
    float inv = 1.0f / (sqrtf(red[0]) + 1e-10f);
    __syncthreads();
#pragma unroll
    for (int i = 0; i < 16; ++i) { int idx = tid + 256 * i; v[idx] = lv[i] * inv; }
    __syncthreads();
  }
  // Rayleigh quotient
#pragma unroll
  for (int i = 0; i < 16; ++i) { int idx = tid + 256 * i; w[idx] = v[idx] * dloc[i]; }
  __syncthreads();
  float dot = 0.0f;
  for (int i = 0; i < 16; ++i) {
    int idx = tid + 256 * i;
    int t = idx >> 5, f = idx & 31;
    float s = w[idx];
    const float* tarow = ta + (h * TT + t) * TT;
    for (int t2 = 0; t2 < TT; ++t2) s += tarow[t2] * w[t2 * FF + f];
    const float* farow = fa + (h * FF + f) * FF;
    const float* wrow = w + t * FF;
    for (int f2 = 0; f2 < FF; ++f2) s += farow[f2] * wrow[f2];
    float lvi = v[idx] - dloc[i] * s;
    dot += v[idx] * lvi;
  }
  red[tid] = dot; __syncthreads();
  for (int st = 128; st > 0; st >>= 1) {
    if (tid < st) red[tid] += red[tid + st];
    __syncthreads();
  }
  if (tid == 0) {
    float sn = fmaxf(fabsf(red[0]), 1.0f);
    cscale[h] = 2.0f / sn;
  }
}

// ---------------- kernel 6: main WMMA kernel ---------------------------------
// grid (B, H), block 256 (8 wave32s). Per workgroup:
//   Y = X .* dinv  (half copies row-major + transposed in LDS)
//   Z = ta^T @ Y + Y @ fa + Y           (WMMA f32<-f16, K=128+32)
//   XL = (c-1)*X - c*Z.*dinv            (f32 epilogue, half into LDS)
//   out = XL @ W1h + xp + bias          (WMMA, K=32)
#define YH_STRIDE 40   // halves; 80B rows (16B aligned)
#define YT_STRIDE 136  // halves; 272B rows
__global__ void main_kernel(const float* __restrict__ x,      // [B][T][F]
                            const float* __restrict__ xp,     // [H][B][T][HD]
                            const _Float16* __restrict__ taT, // [H][T][T]
                            const _Float16* __restrict__ faT, // [H][F][F]
                            const _Float16* __restrict__ W1T, // [H][HD][F]
                            const float* __restrict__ dinv,   // [H][T][F]
                            const float* __restrict__ cscale, // [H]
                            const float* __restrict__ bias,   // [H*HD]
                            float* __restrict__ out) {        // [B][T][H*HD]
  const int b = blockIdx.x, h = blockIdx.y;
  const int tid = threadIdx.x;
  const int lane = tid & 31, wave = tid >> 5;
  const int mrow = lane & 15;       // M/N position within tile
  const int hs = lane >> 4;         // half-wave select

  __shared__ alignas(16) _Float16 Yh[TT * YH_STRIDE];
  __shared__ alignas(16) _Float16 YTs[FF * YT_STRIDE];
  __shared__ alignas(16) _Float16 XLh[TT * YH_STRIDE];

  const float* xb = x + b * TT * FF;
  const float* dv = dinv + h * TT * FF;
  const float c = cscale[h];

#pragma unroll
  for (int i = 0; i < 16; ++i) {
    int idx = tid + 256 * i;
    int t = idx >> 5, f = idx & 31;
    float y = xb[idx] * dv[idx];
    _Float16 yh = (_Float16)y;
    Yh[t * YH_STRIDE + f] = yh;
    YTs[f * YT_STRIDE + t] = yh;
  }
  __syncthreads();

  const _Float16* taTh = taT + h * TT * TT;
  const _Float16* faTh = faT + h * FF * FF;
  const int m = wave * 16 + mrow;          // output row this lane's fragment covers
  const _Float16* arow_ta = taTh + m * TT;

#pragma unroll
  for (int nt = 0; nt < 2; ++nt) {
    v8f acc = {0.f, 0.f, 0.f, 0.f, 0.f, 0.f, 0.f, 0.f};
    // Z1: ta^T @ Y  (K = 128, 4 steps)
#pragma unroll
    for (int kb = 0; kb < 4; ++kb) {
      const int k0 = kb * 32;
      v16h a = make_a_frag(arow_ta + k0 + hs * 16, arow_ta + k0 + hs * 16 + 8);
      v16h bfr = make_b_frag(&YTs[(nt * 16 + mrow) * YT_STRIDE + k0 + hs * 16]);
      acc = __builtin_amdgcn_wmma_f32_16x16x32_f16(false, a, false, bfr, (short)0, acc,
                                                   false, false);
    }
    // Z2: Y @ fa  (K = 32, one step)
    {
      v16h a = make_a_frag(&Yh[m * YH_STRIDE + hs * 16], &Yh[m * YH_STRIDE + hs * 16 + 8]);
      v16h bfr = make_b_frag(faTh + (nt * 16 + mrow) * FF + hs * 16);
      acc = __builtin_amdgcn_wmma_f32_16x16x32_f16(false, a, false, bfr, (short)0, acc,
                                                   false, false);
    }
    // epilogue: Z += Y (f32); XL = (c-1)x - c*Z*dinv -> half into LDS
#pragma unroll
    for (int r = 0; r < 8; ++r) {
      int t = wave * 16 + hs * 8 + r;
      int f = nt * 16 + mrow;
      float xv = xb[t * FF + f];
      float di = dv[t * FF + f];
      float Z = acc[r] + xv * di;
      float XL = (c - 1.0f) * xv - c * Z * di;
      XLh[t * YH_STRIDE + f] = (_Float16)XL;
    }
  }
  __syncthreads();

  // out = XL @ W1h + xp + bias   (M=128, N=16, K=32; one tile per wave)
  {
    v8f acc = {0.f, 0.f, 0.f, 0.f, 0.f, 0.f, 0.f, 0.f};
    v16h a = make_a_frag(&XLh[m * YH_STRIDE + hs * 16], &XLh[m * YH_STRIDE + hs * 16 + 8]);
    v16h bfr = make_b_frag(W1T + (h * HDD + mrow) * FF + hs * 16);
    acc = __builtin_amdgcn_wmma_f32_16x16x32_f16(false, a, false, bfr, (short)0, acc,
                                                 false, false);
#pragma unroll
    for (int r = 0; r < 8; ++r) {
      int t = wave * 16 + hs * 8 + r;
      int j = mrow;
      float h0 = xp[((h * BB + b) * TT + t) * HDD + j];
      out[(b * TT + t) * (HH * HDD) + h * HDD + j] = acc[r] + h0 + bias[h * HDD + j];
    }
  }
}

// ---------------- launcher ---------------------------------------------------
extern "C" void kernel_launch(void* const* d_in, const int* in_sizes, int n_in,
                              void* d_out, int out_size, void* d_ws, size_t ws_size,
                              hipStream_t stream) {
  (void)in_sizes; (void)n_in; (void)out_size;
  const float* x      = (const float*)d_in[0];
  const float* weight = (const float*)d_in[1];
  const float* bias   = (const float*)d_in[2];
  const float* tq     = (const float*)d_in[3];
  const float* tk     = (const float*)d_in[4];
  const float* ff     = (const float*)d_in[5];

  char* p = (char*)d_ws;
  auto alloc = [&](size_t bytes) -> void* {
    void* r = (void*)p;
    p += (bytes + 255) & ~(size_t)255;
    return r;
  };
  float*    fa    = (float*)alloc(sizeof(float) * HH * FF * FF);
  float*    rs_fa = (float*)alloc(sizeof(float) * HH * FF);
  float*    xp    = (float*)alloc(sizeof(float) * HH * BB * TT * HDD);
  float*    q     = (float*)alloc(sizeof(float) * HH * BB * TT);
  float*    kk    = (float*)alloc(sizeof(float) * HH * BB * TT);
  float*    ta    = (float*)alloc(sizeof(float) * HH * TT * TT);
  float*    rs_ta = (float*)alloc(sizeof(float) * HH * TT);
  float*    dinv  = (float*)alloc(sizeof(float) * HH * TT * FF);
  float*    csc   = (float*)alloc(sizeof(float) * HH);
  _Float16* faT   = (_Float16*)alloc(sizeof(_Float16) * HH * FF * FF);
  _Float16* taT   = (_Float16*)alloc(sizeof(_Float16) * HH * TT * TT);
  _Float16* W1T   = (_Float16*)alloc(sizeof(_Float16) * HH * HDD * FF);
  if ((size_t)(p - (char*)d_ws) > ws_size) return;  // workspace too small

  fa_kernel<<<HH, 32, 0, stream>>>(ff, fa, faT, rs_fa);
  w1t_kernel<<<(HH * HDD * FF + 255) / 256, 256, 0, stream>>>(weight, W1T);
  xp_kernel<<<(HH * BB * TT + 127) / 128, 128, 0, stream>>>(x, weight, tq, tk, xp, q, kk);
  ta_kernel<<<dim3(TT, HH), TT, 0, stream>>>(q, kk, ta, taT, rs_ta);
  power_kernel<<<HH, 256, 0, stream>>>(ta, fa, rs_ta, rs_fa, dinv, csc);
  main_kernel<<<dim3(BB, HH), 256, 0, stream>>>(x, xp, taT, faT, W1T, dinv, csc, bias,
                                                (float*)d_out);
}
